// AdaptiveNet_SLSTM_56272661512584
// MI455X (gfx1250) — matrix operand order repaired and checked by reading, not compiled, not run on hardware
//
#include <hip/hip_runtime.h>
#include <hip/hip_bf16.h>
#include <math.h>

// ---------------------------------------------------------------------------
// AdaptiveNet SLSTM forward for MI455X (gfx1250, wave32).
//  - bf16 WMMA (v_wmma_f32_16x16x32_bf16) for the per-scan-step gate GEMMs
//  - Both A and B tiles staged by the Tensor Data Mover (tensor_load_to_lds
//    with LDS padding), double-buffered so DMA overlaps the WMMA stream
//  - Weights pre-packed fragment-ready: B fragment = one 32B LDS read
//  - BatchNorm folded into layer-2 input weights (spikes are binary)
// ---------------------------------------------------------------------------

#define B_    64
#define T_    4096
#define C_    14
#define H_    128
#define G4_   512      // 4*H gates
#define CO_   32
#define NCLS_ 8

#define PSTRIDE 40     // padded LDS row stride (elements) per 32-elem run

typedef __bf16 bf16;
typedef __bf16 v16bf __attribute__((ext_vector_type(16)));
typedef __bf16 v8bf  __attribute__((ext_vector_type(8)));
typedef float  v8f   __attribute__((ext_vector_type(8)));
typedef unsigned int u32x4 __attribute__((ext_vector_type(4)));
typedef int          i32x8 __attribute__((ext_vector_type(8)));
typedef int          i32x4 __attribute__((ext_vector_type(4)));

// ---------------------------------------------------------------------------
// TDM 2D tile load: dim0elems x rows bf16, global row stride `stride` elems,
// with 4 dwords of LDS padding inserted every 16 dwords (one 32-elem run) so
// each run lands at a 40-element (80B) stride -> conflict-free ds_load_b128
// fragment reads. D# per CDNA5 ISA ch.10.8.
// ---------------------------------------------------------------------------
__device__ __forceinline__ void tdm_load_2d(const bf16* gsrc, bf16* lds_dst,
                                            int dim0elems, int rows, int stride)
{
    unsigned lds_addr = (unsigned)(unsigned long long)lds_dst;  // low 32b = LDS offset
    unsigned long long ga = (unsigned long long)gsrc;
    u32x4 g0;
    g0.x = 1u;                                                  // count=1, user mode
    g0.y = lds_addr;                                            // bits 63:32  lds_addr
    g0.z = (unsigned)(ga & 0xffffffffull);                      // bits 95:64  global lo
    g0.w = (unsigned)((ga >> 32) & 0x1ffffffull) | (2u << 30);  // addr hi, type=2
    i32x8 g1;
    // data_size=1 (2B) | pad_enable | pad_interval=3 (16 dwords) | pad_amount=3 (4 dwords)
    g1[0] = (1 << 16) | (1 << 20) | (3 << 22) | (3 << 25);
    g1[1] = (dim0elems & 0xffff) << 16;  // tensor_dim0 low16
    g1[2] = (rows & 0xffff) << 16;       // tensor_dim0 hi=0, tensor_dim1 low16
    g1[3] = (dim0elems & 0xffff) << 16;  // tensor_dim1 hi=0, tile_dim0
    g1[4] = rows & 0xffff;               // tile_dim1 = rows, tile_dim2 = 0
    g1[5] = stride;                      // tensor_dim0_stride low32 (elements)
    g1[6] = 0;
    g1[7] = 0;
    i32x4 z4 = {0, 0, 0, 0};
    i32x8 z8 = {0, 0, 0, 0, 0, 0, 0, 0};
    __builtin_amdgcn_tensor_load_to_lds(g0, g1, z4, z4, z8, 0);
}

// ---------------------------------------------------------------------------
// gates[4096,512] = Ax[4096,Kx]@Wx + Ah[4096,128]@Wh + bias
// Wx/Wh pre-packed [kchunk][512][32] bf16 (fragment-ready, see k_w_pack).
// Block = 256 threads = 8 waves; block tile 32(M) x 256(N); wave (wm,wn)
// owns a 16x64 strip = 4 WMMA accumulators with A-fragment reuse.
// Double-buffered: chunk c+1 staged (two TDM descriptors, wave 0 only)
// during compute of chunk c. No per-thread global loads in the loop.
// ---------------------------------------------------------------------------
__global__ void __launch_bounds__(256)
k_gates_gemm(const bf16* __restrict__ Ax, int Kx, const bf16* __restrict__ Bx,
             const bf16* __restrict__ Ah, const bf16* __restrict__ Bh,
             const float* __restrict__ bias, float* __restrict__ G)
{
    __shared__ __align__(16) bf16 sA[2][32 * PSTRIDE];
    __shared__ __align__(16) bf16 sB[2][256 * PSTRIDE];

    const int m0 = blockIdx.y * 32, n0 = blockIdx.x * 256;
    const int nx  = Kx >> 5;                 // x-part chunks
    const int nch = nx + (H_ >> 5);          // + recurrent chunks
    const int tid = threadIdx.x, wave = tid >> 5, lane = tid & 31;
    const int half = lane >> 4, lr = lane & 15, wm = wave >> 2, wn = wave & 3;

    auto stage = [&](int c, int buf) {       // wave 0 only: issue two TDM copies
        const bf16* A; int lda, k0; const bf16* Bp;
        if (c < nx) { A = Ax; lda = Kx; k0 = c << 5;
                      Bp = Bx + ((size_t)c * G4_ + n0) * 32; }
        else        { int ch = c - nx; A = Ah; lda = H_; k0 = ch << 5;
                      Bp = Bh + ((size_t)ch * G4_ + n0) * 32; }
        tdm_load_2d(A + (size_t)m0 * lda + k0, &sA[buf][0], 32, 32, lda);
        tdm_load_2d(Bp, &sB[buf][0], 8192, 1, 8192);
    };

    v8f acc[4] = {};
    if (wave == 0) stage(0, 0);
    for (int c = 0; c < nch; ++c) {
        const int buf = c & 1;
        if (wave == 0) __builtin_amdgcn_s_wait_tensorcnt(0);
        __syncthreads();                     // publish chunk c; chunk c-1 consumed
        if (wave == 0 && c + 1 < nch) stage(c + 1, buf ^ 1);

        // A fragment: row m = wm*16+lr; lane half selects K-run {0..15}
        // or {16..31} -> two 16B LDS reads at 80B row stride.
        const bf16* pa = &sA[buf][(wm * 16 + lr) * PSTRIDE + half * 16];
        v8bf alo = *(const v8bf*)pa;
        v8bf ahi = *(const v8bf*)(pa + 8);
        v16bf a = __builtin_shufflevector(alo, ahi,
                    0, 1, 2, 3, 4, 5, 6, 7, 8, 9, 10, 11, 12, 13, 14, 15);

        // Prefetch all four B fragments, then issue the WMMAs back-to-back.
        v16bf bf[4];
#pragma unroll
        for (int j = 0; j < 4; ++j) {
            const bf16* pb = &sB[buf][(wn * 64 + j * 16 + lr) * PSTRIDE + half * 16];
            bf[j] = *(const v16bf*)pb;
        }
#pragma unroll
        for (int j = 0; j < 4; ++j)
            acc[j] = __builtin_amdgcn_wmma_f32_16x16x32_bf16(
                false, a, false, bf[j], (short)0, acc[j], false, false);
    }

#pragma unroll
    for (int j = 0; j < 4; ++j)
#pragma unroll
        for (int v = 0; v < 8; ++v) {
            int gm = m0 + wm * 16 + half * 8 + v;    // C/D layout: vgpr v -> row
            int gn = n0 + wn * 64 + j * 16 + lr;
            G[(size_t)gm * G4_ + gn] = acc[j][v] + bias[gn];
        }
}

// ---------------------------------------------------------------------------
// LSTM cell elementwise update with snnTorch subtract-reset + fast-sigmoid spike
// ---------------------------------------------------------------------------
__device__ __forceinline__ float sigmoidf_(float x) { return 1.f / (1.f + __expf(-x)); }

__global__ void k_lstm_update(const float* __restrict__ G,
                              float* __restrict__ c, float* __restrict__ h,
                              bf16* __restrict__ hbf,
                              const float* __restrict__ thr_p,
                              bf16* __restrict__ spk_out,     // layer1: spikes out
                              float* __restrict__ mem_acc)    // layer2: mean accumulator
{
    int idx = blockIdx.x * blockDim.x + threadIdx.x;
    if (idx >= T_ * H_) return;
    int t = idx >> 7, hh = idx & (H_ - 1);
    const float* g = G + (size_t)t * G4_;
    float thr = thr_p[0];
    float gi = sigmoidf_(g[hh]);
    float gf = sigmoidf_(g[H_ + hh]);
    float gg = tanhf(g[2 * H_ + hh]);
    float go = sigmoidf_(g[3 * H_ + hh]);
    float hprev = h[idx];
    float reset = (hprev > thr) ? thr : 0.f;   // detached Heaviside * thr
    float cn = gf * c[idx] + gi * gg;
    float hn = go * tanhf(cn) - reset;
    c[idx] = cn;
    h[idx] = hn;
    hbf[idx] = (bf16)hn;                       // bf16 copy feeds next step's WMMA
    if (spk_out) spk_out[idx] = (bf16)((hn - thr > 0.f) ? 1.f : 0.f);
    if (mem_acc) mem_acc[idx] += hn;
}

// ---------------------------------------------------------------------------
// Stage 1: adaptive delta modulation. One block per (b,c); unbiased std over T-1.
// ---------------------------------------------------------------------------
__global__ void k_delta(const float* __restrict__ x, bf16* __restrict__ spk_in)
{
    __shared__ float s1[256], s2[256];
    int b = blockIdx.x / C_, c = blockIdx.x % C_;
    const float* xb = x + (size_t)b * T_ * C_ + c;
    float sum = 0.f, sq = 0.f;
    for (int t = threadIdx.x; t < T_ - 1; t += 256) {
        float d = xb[(t + 1) * C_] - xb[t * C_];
        sum += d; sq += d * d;
    }
    s1[threadIdx.x] = sum; s2[threadIdx.x] = sq;
    __syncthreads();
    for (int off = 128; off > 0; off >>= 1) {
        if (threadIdx.x < off) {
            s1[threadIdx.x] += s1[threadIdx.x + off];
            s2[threadIdx.x] += s2[threadIdx.x + off];
        }
        __syncthreads();
    }
    float n = (float)(T_ - 1);
    float mean = s1[0] / n;
    float var  = (s2[0] - s1[0] * s1[0] / n) / (n - 1.f);
    float athr = mean + 2.5f * sqrtf(fmaxf(var, 0.f));
    bf16* ob = spk_in + (size_t)b * T_ * C_ + c;
    if (threadIdx.x == 0) ob[0] = (bf16)0.f;
    for (int t = threadIdx.x; t < T_ - 1; t += 256) {
        float d = xb[(t + 1) * C_] - xb[t * C_];
        ob[(t + 1) * C_] = (bf16)((fabsf(d) > athr) ? 1.f : 0.f);
    }
}

// Stage 2: conv1d(k=3, SAME) over time + Leaky spike at threshold 1.0
__global__ void k_conv_spike(const bf16* __restrict__ spk_in,
                             const float* __restrict__ cw, const float* __restrict__ cb,
                             bf16* __restrict__ cur1)
{
    int idx = blockIdx.x * 256 + threadIdx.x;
    if (idx >= B_ * T_ * CO_) return;
    int o = idx % CO_;
    int t = (idx / CO_) % T_;
    int b = idx / (CO_ * T_);
    float acc = cb[o];
    const bf16* sb = spk_in + (size_t)b * T_ * C_;
#pragma unroll
    for (int k = 0; k < 3; ++k) {
        int tt = t + k - 1;
        if (tt < 0 || tt >= T_) continue;
        for (int c = 0; c < C_; ++c)
            acc += cw[(o * C_ + c) * 3 + k] * (float)sb[tt * C_ + c];
    }
    cur1[idx] = (bf16)((acc - 1.f > 0.f) ? 1.f : 0.f);   // spike_atan fwd = Heaviside
}

// Weight prep: W[512,K] fp32 -> packed bf16 [kchunk][512][32] (fragment-ready)
__global__ void k_w_pack(const float* __restrict__ W, bf16* __restrict__ out, int K)
{
    int idx = blockIdx.x * 256 + threadIdx.x;
    if (idx >= K * G4_) return;
    int k = idx / G4_, n = idx % G4_;
    out[((size_t)(k >> 5) * G4_ + n) * 32 + (k & 31)] = (bf16)W[n * K + k];
}

__global__ void k_bias_sum(const float* __restrict__ a, const float* __restrict__ b,
                           float* __restrict__ out)
{
    int n = blockIdx.x * 256 + threadIdx.x;
    if (n < G4_) out[n] = a[n] + b[n];
}

// BN stats: spikes are binary => mean = p, var = p(1-p). Count ones per channel.
__global__ void k_bn_count(const bf16* __restrict__ spk1, float* __restrict__ cnt)
{
    __shared__ float sh[128];
    int tid = threadIdx.x;
    if (tid < 128) sh[tid] = 0.f;
    __syncthreads();
    float local = 0.f;
    size_t total = (size_t)B_ * T_ * H_;
    for (size_t i = blockIdx.x * 256ull + tid; i < total; i += (size_t)gridDim.x * 256ull)
        local += (float)spk1[i];               // stride multiple of 128 -> channel fixed
    atomicAdd(&sh[tid & 127], local);
    __syncthreads();
    if (tid < 128) atomicAdd(&cnt[tid], sh[tid]);
}

__global__ void k_bn_fin(const float* __restrict__ cnt, const float* __restrict__ gamma,
                         const float* __restrict__ beta,
                         float* __restrict__ bna, float* __restrict__ bnb)
{
    int hh = threadIdx.x;
    if (hh < H_) {
        float p = cnt[hh] * (1.f / (float)((size_t)B_ * T_));
        float var = p - p * p;                 // biased var of a Bernoulli
        float a = gamma[hh] * rsqrtf(var + 1e-5f);
        bna[hh] = a;
        bnb[hh] = beta[hh] - p * a;
    }
}

// Fold BN scale into layer-2 input weights (packed layout): Wx2 = a_k * w_ih2[n,k]
__global__ void k_w2_scaled(const float* __restrict__ w_ih2, const float* __restrict__ bna,
                            bf16* __restrict__ out)
{
    int idx = blockIdx.x * 256 + threadIdx.x;
    if (idx >= H_ * G4_) return;
    int k = idx / G4_, n = idx % G4_;
    out[((size_t)(k >> 5) * G4_ + n) * 32 + (k & 31)] = (bf16)(w_ih2[n * H_ + k] * bna[k]);
}

// bias2[n] = b_ih2 + b_hh2 + sum_k bnb_k * w_ih2[n,k]  (BN shift folded)
__global__ void k_bias2(const float* __restrict__ b_ih2, const float* __restrict__ b_hh2,
                        const float* __restrict__ w_ih2, const float* __restrict__ bnb,
                        float* __restrict__ out)
{
    int n = blockIdx.x * 256 + threadIdx.x;
    if (n >= G4_) return;
    float s = b_ih2[n] + b_hh2[n];
    for (int k = 0; k < H_; ++k) s += bnb[k] * w_ih2[n * H_ + k];
    out[n] = s;
}

__global__ void k_zero(unsigned* __restrict__ p, size_t n)
{
    for (size_t i = blockIdx.x * 256ull + threadIdx.x; i < n; i += (size_t)gridDim.x * 256ull)
        p[i] = 0u;
}

// out[t,n] = (facc[t,:]/64) . fc_w[n,:] + fc_b[n]
__global__ void k_fc(const float* __restrict__ facc, const float* __restrict__ fw,
                     const float* __restrict__ fb, float* __restrict__ out)
{
    int idx = blockIdx.x * 256 + threadIdx.x;
    if (idx >= T_ * NCLS_) return;
    int n = idx & 7, t = idx >> 3;
    float s = 0.f;
    for (int hh = 0; hh < H_; ++hh) s += facc[t * H_ + hh] * fw[n * H_ + hh];
    out[idx] = s * (1.f / 64.f) + fb[n];
}

// ---------------------------------------------------------------------------
extern "C" void kernel_launch(void* const* d_in, const int* in_sizes, int n_in,
                              void* d_out, int out_size, void* d_ws, size_t ws_size,
                              hipStream_t stream)
{
    const float* x      = (const float*)d_in[0];
    const float* conv_w = (const float*)d_in[1];
    const float* conv_b = (const float*)d_in[2];
    const float* w_ih1  = (const float*)d_in[3];
    const float* w_hh1  = (const float*)d_in[4];
    const float* b_ih1  = (const float*)d_in[5];
    const float* b_hh1  = (const float*)d_in[6];
    const float* thr1   = (const float*)d_in[7];
    const float* w_ih2  = (const float*)d_in[8];
    const float* w_hh2  = (const float*)d_in[9];
    const float* b_ih2  = (const float*)d_in[10];
    const float* b_hh2  = (const float*)d_in[11];
    const float* thr2   = (const float*)d_in[12];
    const float* gamma  = (const float*)d_in[13];
    const float* beta   = (const float*)d_in[14];
    const float* fc_w   = (const float*)d_in[15];
    const float* fc_b   = (const float*)d_in[16];
    float* out = (float*)d_out;
    (void)in_sizes; (void)n_in; (void)out_size; (void)ws_size;

    char* p = (char*)d_ws;
    auto carve = [&](size_t bytes) { char* r = p; p += (bytes + 255) & ~(size_t)255; return r; };
    bf16*  spk_in = (bf16*)carve((size_t)B_ * T_ * C_ * 2);   //  7.0 MB
    bf16*  cur1   = (bf16*)carve((size_t)B_ * T_ * CO_ * 2);  // 16   MB
    bf16*  spk1   = (bf16*)carve((size_t)B_ * T_ * H_ * 2);   // 64   MB
    float* G      = (float*)carve((size_t)T_ * G4_ * 4);      //  8   MB
    float* cst    = (float*)carve((size_t)T_ * H_ * 4);
    float* hst    = (float*)carve((size_t)T_ * H_ * 4);
    bf16*  hbf    = (bf16*)carve((size_t)T_ * H_ * 2);
    float* facc   = (float*)carve((size_t)T_ * H_ * 4);
    bf16*  Wx1    = (bf16*)carve((size_t)CO_ * G4_ * 2);
    bf16*  Wh1    = (bf16*)carve((size_t)H_ * G4_ * 2);
    bf16*  Wx2    = (bf16*)carve((size_t)H_ * G4_ * 2);
    bf16*  Wh2    = (bf16*)carve((size_t)H_ * G4_ * 2);
    float* bias1  = (float*)carve(G4_ * 4);
    float* bias2  = (float*)carve(G4_ * 4);
    float* bncnt  = (float*)carve(H_ * 4);
    float* bna    = (float*)carve(H_ * 4);
    float* bnb    = (float*)carve(H_ * 4);

    // Stage 1+2: delta spikes, conv+spike
    k_delta<<<B_ * C_, 256, 0, stream>>>(x, spk_in);
    k_conv_spike<<<(B_ * T_ * CO_ + 255) / 256, 256, 0, stream>>>(spk_in, conv_w, conv_b, cur1);

    // Weight prep (bf16, fragment-ready packed layout for the WMMA B-operand)
    k_w_pack<<<(CO_ * G4_ + 255) / 256, 256, 0, stream>>>(w_ih1, Wx1, CO_);
    k_w_pack<<<(H_ * G4_ + 255) / 256, 256, 0, stream>>>(w_hh1, Wh1, H_);
    k_w_pack<<<(H_ * G4_ + 255) / 256, 256, 0, stream>>>(w_hh2, Wh2, H_);
    k_bias_sum<<<2, 256, 0, stream>>>(b_ih1, b_hh1, bias1);

    // Zero states (workspace is poisoned by harness)
    k_zero<<<256, 256, 0, stream>>>((unsigned*)cst, (size_t)T_ * H_);
    k_zero<<<256, 256, 0, stream>>>((unsigned*)hst, (size_t)T_ * H_);
    k_zero<<<256, 256, 0, stream>>>((unsigned*)hbf, (size_t)T_ * H_ / 2);
    k_zero<<<1, 128, 0, stream>>>((unsigned*)bncnt, H_);

    // Layer-1 scan over B steps (sequential recurrence)
    for (int s = 0; s < B_; ++s) {
        k_gates_gemm<<<dim3(2, 128), 256, 0, stream>>>(
            cur1 + (size_t)s * T_ * CO_, CO_, Wx1, hbf, Wh1, bias1, G);
        k_lstm_update<<<(T_ * H_ + 255) / 256, 256, 0, stream>>>(
            G, cst, hst, hbf, thr1, spk1 + (size_t)s * T_ * H_, nullptr);
    }

    // BN stats (binary spikes => popcount) and fold into layer-2 input weights
    k_bn_count<<<2048, 256, 0, stream>>>(spk1, bncnt);
    k_bn_fin<<<1, 128, 0, stream>>>(bncnt, gamma, beta, bna, bnb);
    k_w2_scaled<<<(H_ * G4_ + 255) / 256, 256, 0, stream>>>(w_ih2, bna, Wx2);
    k_bias2<<<2, 256, 0, stream>>>(b_ih2, b_hh2, w_ih2, bnb, bias2);

    // Reset states + mean accumulator
    k_zero<<<256, 256, 0, stream>>>((unsigned*)cst, (size_t)T_ * H_);
    k_zero<<<256, 256, 0, stream>>>((unsigned*)hst, (size_t)T_ * H_);
    k_zero<<<256, 256, 0, stream>>>((unsigned*)hbf, (size_t)T_ * H_ / 2);
    k_zero<<<256, 256, 0, stream>>>((unsigned*)facc, (size_t)T_ * H_);

    // Layer-2 scan; record running mean of mem
    for (int s = 0; s < B_; ++s) {
        k_gates_gemm<<<dim3(2, 128), 256, 0, stream>>>(
            spk1 + (size_t)s * T_ * H_, H_, Wx2, hbf, Wh2, bias2, G);
        k_lstm_update<<<(T_ * H_ + 255) / 256, 256, 0, stream>>>(
            G, cst, hst, hbf, thr2, nullptr, facc);
    }

    // FC head
    k_fc<<<(T_ * NCLS_ + 255) / 256, 256, 0, stream>>>(facc, fc_w, fc_b, out);
}
